// IPLCNNFeatureExtractor_6347961663551
// MI455X (gfx1250) — matrix operand order repaired and checked by reading, not compiled, not run on hardware
//
#include <hip/hip_runtime.h>
#include <hip/hip_bf16.h>
#include <math.h>

// CDNA5 (gfx1250) WMMA types
typedef __attribute__((ext_vector_type(16))) __bf16 v16bf;
typedef __attribute__((ext_vector_type(8)))  float  v8f;
typedef __attribute__((ext_vector_type(4)))  unsigned int v4u;
typedef __attribute__((ext_vector_type(8)))  int v8i;
typedef __attribute__((ext_vector_type(4)))  int v4i;

#define BATCH 8
#define BN_EPS 1e-5f

#ifndef __has_builtin
#define __has_builtin(x) 0
#endif
#if __has_builtin(__builtin_amdgcn_tensor_load_to_lds) && __has_builtin(__builtin_amdgcn_s_wait_tensorcnt)
#define HAVE_TDM 1
#else
#define HAVE_TDM 0
#endif

// ---------------------------------------------------------------------------
// Input conversion: x (B,3,256,256) f32 NCHW -> act (B,256,256,32) bf16 NHWC
// ---------------------------------------------------------------------------
__global__ void cvt_in_kernel(const float* __restrict__ x, __bf16* __restrict__ act)
{
    int idx = blockIdx.x * blockDim.x + threadIdx.x;
    const int total = BATCH * 256 * 256;
    if (idx >= total) return;
    int xw = idx & 255;
    int t  = idx >> 8;
    int yh = t & 255;
    int b  = t >> 8;
    __bf16* o = act + (size_t)idx * 32;
    #pragma unroll
    for (int c = 0; c < 32; ++c) {
        float v = 0.f;
        if (c < 3) v = x[(((size_t)b * 3 + c) * 256 + yh) * 256 + xw];
        o[c] = (__bf16)v;
    }
}

// ---------------------------------------------------------------------------
// Weight prep: fold BN scale, convert to bf16, relayout OIHW -> [tap][co][ci]
// ---------------------------------------------------------------------------
__global__ void prep_w_kernel(const float* __restrict__ w,  const float* __restrict__ cb,
                              const float* __restrict__ g,  const float* __restrict__ be,
                              const float* __restrict__ mu, const float* __restrict__ va,
                              __bf16* __restrict__ wt, float* __restrict__ bias2,
                              int Cin, int Cpad, int Cout)
{
    int idx = blockIdx.x * blockDim.x + threadIdx.x;
    int total = 9 * Cout * Cpad;
    if (idx >= total) return;
    int ci  = idx % Cpad;
    int r   = idx / Cpad;
    int co  = r % Cout;
    int tap = r / Cout;
    float inv = g[co] * rsqrtf(va[co] + BN_EPS);
    float val = 0.f;
    if (ci < Cin) {
        int ty = tap / 3, tx = tap % 3;
        val = w[(((size_t)co * Cin + ci) * 3 + ty) * 3 + tx] * inv;
    }
    wt[((size_t)tap * Cout + co) * Cpad + ci] = (__bf16)val;
    if (tap == 0 && ci == 0) bias2[co] = cb[co] * inv + be[co] - mu[co] * inv;
}

// ---------------------------------------------------------------------------
// TDM helper: async DMA of a 64-row x KC-col bf16 weight tile into LDS with
// 16B per-row padding (bank de-conflict). D# per CDNA5 ISA ch.8.
// ---------------------------------------------------------------------------
#if HAVE_TDM
__device__ __forceinline__ void tdm_load_wtile(const __bf16* gsrc, unsigned ldsOff,
                                               int Cpad, int KC, int padi, int rows9)
{
    unsigned long long ga = (unsigned long long)(size_t)gsrc;
    v4u g0;
    g0.x = 1u;                                   // count=1 (valid user descriptor)
    g0.y = ldsOff;                               // LDS byte address
    g0.z = (unsigned)ga;                         // global_addr[31:0]
    g0.w = (unsigned)(ga >> 32) | 0x80000000u;   // global_addr[56:32] | type=2<<30
    v8i g1;
    g1[0] = (1 << 16)        // data_size = 1 -> 2 bytes
          | (1 << 20)        // pad_enable
          | (padi << 22)     // pad_interval: 1<<padi 8B units = KC elements
          | (3 << 25);       // pad_amount code 3 -> 4 DWORDs = 16 bytes
    g1[1] = (Cpad & 0xFFFF) << 16;                               // tensor_dim0 lo16
    g1[2] = ((Cpad >> 16) & 0xFFFF) | ((rows9 & 0xFFFF) << 16);  // dim0 hi | dim1 lo
    g1[3] = ((rows9 >> 16) & 0xFFFF) | ((KC & 0xFFFF) << 16);    // dim1 hi | tile_dim0
    g1[4] = 64;                                                  // tile_dim1=64, tile_dim2=0
    g1[5] = Cpad;                                                // tensor_dim0_stride lo32
    g1[6] = 0;
    g1[7] = 0;
    v4i gz = {0, 0, 0, 0};
#if defined(__clang_major__) && __clang_major__ >= 23
    v8i gz8 = {0, 0, 0, 0, 0, 0, 0, 0};
    __builtin_amdgcn_tensor_load_to_lds(g0, g1, gz, gz, gz8, 0);
#else
    __builtin_amdgcn_tensor_load_to_lds(g0, g1, gz, gz, 0);
#endif
}
#endif

// ---------------------------------------------------------------------------
// Implicit-GEMM conv3x3 (pad 1) + fused BN + ReLU via v_wmma_f32_16x16x32_bf16.
// Template constants: all index math folds to shifts/masks, kc loop fully
// unrolls. Workgroup = 8 waves sharing a 64-wide cout block (weights staged
// once per chunk via TDM into LDS); each wave owns a 16-pixel tile and
// computes a 16x64 tile with 4 accumulators.
// Per-lane A fragment = TWO contiguous 16B spans (contiguous k order used here; B matches):
//   elements 0..7  -> K = khalf*16 + 0..7       (one global_load_b128)
//   elements 8..15 -> K = khalf*16 + 8..15      (one global_load_b128)
// ---------------------------------------------------------------------------
template<int H, int W, int CPAD, int COUT>
__global__ void __launch_bounds__(256) conv_wmma_kernel(
    const __bf16* __restrict__ act, const __bf16* __restrict__ wt,
    const float* __restrict__ bias, float* __restrict__ hout)
{
    constexpr int HW   = H * W;
    constexpr int KC   = (CPAD < 256) ? CPAD : 256;   // K chunk staged in LDS
    constexpr int ROWB = KC * 2 + 16;                 // padded LDS row stride
    constexpr int KCN  = KC >> 5;                     // 16x16x32 steps per chunk
    constexpr int NBLK = COUT >> 6;                   // 64-wide cout blocks
    constexpr int PADI = (KC == 512) ? 7 : (KC == 256) ? 6 : (KC == 128) ? 5
                       : (KC == 64) ? 4 : 3;          // interval = KC elements
    extern __shared__ char smem[];

    const int lane = threadIdx.x & 31;
    const int wave = threadIdx.x >> 5;

    const int wgN = blockIdx.x % NBLK;                // pow2 -> mask
    const int wgM = blockIdx.x / NBLK;                // pow2 -> shift
    const int co0 = wgN << 6;

    const int pg0 = (wgM * 8 + wave) << 4;            // global pixel of this tile
    const int b   = pg0 / HW;
    const int p0  = pg0 - b * HW;

    const int mloc  = lane & 15;
    const int khalf = lane >> 4;
    const int p  = p0 + mloc;
    const int oy = p / W;
    const int ox = p % W;

    const unsigned ldsOff = (unsigned)__builtin_amdgcn_groupstaticsize();

    v8f acc0 = {}, acc1 = {}, acc2 = {}, acc3 = {};

    for (int tap = 0; tap < 9; ++tap) {
        const int dy = tap / 3 - 1;
        const int dx = tap % 3 - 1;
        const int iy = oy + dy;
        const int ix = ox + dx;
        const bool valid = ((unsigned)iy < (unsigned)H) && ((unsigned)ix < (unsigned)W);
        const __bf16* abase = act + ((size_t)((b * H + iy) * W + ix)) * CPAD + khalf * 16;
        const __bf16* wrow  = wt  + ((size_t)(tap * COUT + co0)) * CPAD;
        if (tap < 8)   // L2 prefetch of next tap's weight rows -> global_prefetch
            __builtin_prefetch(wt + ((size_t)((tap + 1) * COUT + co0)) * CPAD, 0, 1);

        for (int kb = 0; kb < CPAD; kb += KC) {
            __syncthreads();                     // prior chunk's LDS reads done
#if HAVE_TDM
            if (wave == 0) {
                tdm_load_wtile(wrow + kb, ldsOff, CPAD, KC, PADI, 9 * COUT);
                __builtin_amdgcn_s_wait_tensorcnt(0);
            }
#else
            {   // cooperative copy, identical padded layout
                constexpr int CPR = (KC * 2) >> 4;         // 16B chunks per row
                constexpr int TOT = CPR << 6;              // 64 rows
                for (int c = threadIdx.x; c < TOT; c += 256) {
                    int rrow = c / CPR;
                    int off  = c - rrow * CPR;
                    *(uint4*)(smem + rrow * ROWB + (off << 4)) =
                        *(const uint4*)((const char*)(wrow + (size_t)rrow * CPAD + kb) + (off << 4));
                }
            }
#endif
            __syncthreads();                     // weights visible to all waves

            #pragma unroll
            for (int kc = 0; kc < KCN; ++kc) {
                // A fragment: two contiguous b128 loads per lane
                union { v16bf v; uint4 q[2]; } a;
                const __bf16* ap = abase + kb + kc * 32;
                a.q[0] = valid ? *(const uint4*)(ap)      : (uint4){0, 0, 0, 0};
                a.q[1] = valid ? *(const uint4*)(ap + 8) : (uint4){0, 0, 0, 0};
                // B fragments: 4 x (2 x ds_load_b128) from staged LDS tile
                const char* bbase = smem + mloc * ROWB + (kc * 32 + khalf * 16) * 2;
                union { v16bf v; uint4 q[2]; } b0, b1, b2, b3;
                b0.q[0] = *(const uint4*)(bbase);
                b0.q[1] = *(const uint4*)(bbase + 16);
                b1.q[0] = *(const uint4*)(bbase + 16 * ROWB);
                b1.q[1] = *(const uint4*)(bbase + 16 * ROWB + 16);
                b2.q[0] = *(const uint4*)(bbase + 32 * ROWB);
                b2.q[1] = *(const uint4*)(bbase + 32 * ROWB + 16);
                b3.q[0] = *(const uint4*)(bbase + 48 * ROWB);
                b3.q[1] = *(const uint4*)(bbase + 48 * ROWB + 16);
                acc0 = __builtin_amdgcn_wmma_f32_16x16x32_bf16(false, a.v, false, b0.v, (short)0, acc0, false, false);
                acc1 = __builtin_amdgcn_wmma_f32_16x16x32_bf16(false, a.v, false, b1.v, (short)0, acc1, false, false);
                acc2 = __builtin_amdgcn_wmma_f32_16x16x32_bf16(false, a.v, false, b2.v, (short)0, acc2, false, false);
                acc3 = __builtin_amdgcn_wmma_f32_16x16x32_bf16(false, a.v, false, b3.v, (short)0, acc3, false, false);
            }
        }
    }

    // epilogue: bias + ReLU, f32 NHWC stores
    const float bv0 = bias[co0 + mloc];
    const float bv1 = bias[co0 + 16 + mloc];
    const float bv2 = bias[co0 + 32 + mloc];
    const float bv3 = bias[co0 + 48 + mloc];
    #pragma unroll
    for (int r = 0; r < 8; ++r) {
        const int m  = r + khalf * 8;
        const int pp = p0 + m;
        const int y = pp / W, x = pp % W;
        float* orow = hout + ((size_t)((b * H + y) * W + x)) * COUT + co0 + mloc;
        float v0 = acc0[r] + bv0;
        float v1 = acc1[r] + bv1;
        float v2 = acc2[r] + bv2;
        float v3 = acc3[r] + bv3;
        orow[0]  = v0 > 0.f ? v0 : 0.f;
        orow[16] = v1 > 0.f ? v1 : 0.f;
        orow[32] = v2 > 0.f ? v2 : 0.f;
        orow[48] = v3 > 0.f ? v3 : 0.f;
    }
}

// ---------------------------------------------------------------------------
// Fused maxpool2x2 + IPL update (exp-weighted positional pooling)
// ---------------------------------------------------------------------------
template<int H, int W, int COUT>
__global__ void pool_ipl_kernel(
    const float* __restrict__ h, const float* __restrict__ pxin, const float* __restrict__ pyin,
    __bf16* __restrict__ actout, float* __restrict__ pxout, float* __restrict__ pyout,
    int Cprev, int shift, int firstBlock)
{
    constexpr int Ho = H >> 1, Wo = W >> 1;
    long idx = (long)blockIdx.x * blockDim.x + threadIdx.x;
    const long total = (long)BATCH * Ho * Wo * COUT;
    if (idx >= total) return;
    int co = (int)(idx % COUT);
    long r = idx / COUT;
    int x2 = (int)(r % Wo); r /= Wo;
    int y2 = (int)(r % Ho);
    int b  = (int)(r / Ho);

    float hv[4];
    float m = -3.4e38f;
    #pragma unroll
    for (int t = 0; t < 4; ++t) {
        int y = 2 * y2 + (t >> 1);
        int x = 2 * x2 + (t & 1);
        hv[t] = h[((size_t)((b * H + y) * W + x)) * COUT + co];
        m = fmaxf(m, hv[t]);
    }
    float s = 0.f, sx = 0.f, sy = 0.f;
    #pragma unroll
    for (int t = 0; t < 4; ++t) {
        int y = 2 * y2 + (t >> 1);
        int x = 2 * x2 + (t & 1);
        float e = __expf(hv[t] - m);
        float pxv, pyv;
        if (firstBlock) {
            pxv = -1.f + 2.f * (float)y / (float)(H - 1);   // linspace over rows
            pyv = -1.f + 2.f * (float)x / (float)(W - 1);   // linspace over cols
        } else {
            size_t base = ((size_t)((b * H + y) * W + x)) * Cprev + (co >> shift);
            pxv = pxin[base];
            pyv = pyin[base];
        }
        s += e; sx += e * pxv; sy += e * pyv;
    }
    size_t o = ((size_t)((b * Ho + y2) * Wo + x2)) * COUT + co;
    actout[o] = (__bf16)m;
    pxout[o] = sx / s;
    pyout[o] = sy / s;
}

// ---------------------------------------------------------------------------
// Head
// ---------------------------------------------------------------------------
__global__ void head1_kernel(const float* __restrict__ px, const float* __restrict__ py,
                             const float* __restrict__ lw, const float* __restrict__ lb,
                             float* __restrict__ t)
{
    int idx = blockIdx.x * blockDim.x + threadIdx.x;
    if (idx >= BATCH * 512) return;
    int o = idx & 511, b = idx >> 9;
    const float* w0 = lw + (size_t)o * 1024;
    float acc = 0.f;
    for (int hw = 0; hw < 16; ++hw) {
        const float* pxp = px + ((size_t)b * 16 + hw) * 512;
        const float* pyp = py + ((size_t)b * 16 + hw) * 512;
        for (int c = 0; c < 512; ++c)
            acc += w0[c] * pxp[c] + w0[512 + c] * pyp[c];
    }
    t[(size_t)b * 512 + o] = acc * (1.f / 16.f) + lb[o];
}

__global__ void head2_kernel(const float* __restrict__ t, const float* __restrict__ fcw,
                             const float* __restrict__ fcb, float* __restrict__ out)
{
    __shared__ float red[256];
    int b = blockIdx.x;
    float s = 0.f;
    for (int o = threadIdx.x; o < 512; o += blockDim.x)
        s += fcw[o] * t[(size_t)b * 512 + o];
    red[threadIdx.x] = s;
    __syncthreads();
    for (int st = blockDim.x >> 1; st > 0; st >>= 1) {
        if ((int)threadIdx.x < st) red[threadIdx.x] += red[threadIdx.x + st];
        __syncthreads();
    }
    if (threadIdx.x == 0) out[b] = red[0] + fcb[0];
}

// ---------------------------------------------------------------------------
// Launch. d_in order (setup_inputs dict, insertion order, lists flattened):
//  [0] x, [1..6] conv_w, [7..12] conv_b, [13..18] gamma, [19..24] beta,
//  [25..30] mean, [31..36] var, [37] last_w, [38] last_b, [39] fc_w, [40] fc_b
// ---------------------------------------------------------------------------
extern "C" void kernel_launch(void* const* d_in, const int* in_sizes, int n_in,
                              void* d_out, int out_size, void* d_ws, size_t ws_size,
                              hipStream_t stream)
{
    (void)in_sizes; (void)n_in; (void)out_size; (void)ws_size;

    static const int CIN[6]   = {3, 64, 128, 256, 512, 512};
    static const int CPAD[6]  = {32, 64, 128, 256, 512, 512};
    static const int COUT[6]  = {64, 128, 256, 512, 512, 512};
    static const int HIN[6]   = {256, 128, 64, 32, 16, 8};
    static const int SHIFT[6] = {0, 1, 1, 1, 0, 0};

    const float* x = (const float*)d_in[0];
    const float *cw[6], *cb[6], *gm[6], *bt[6], *mu[6], *va[6];
    for (int i = 0; i < 6; ++i) {
        cw[i] = (const float*)d_in[1 + i];
        cb[i] = (const float*)d_in[7 + i];
        gm[i] = (const float*)d_in[13 + i];
        bt[i] = (const float*)d_in[19 + i];
        mu[i] = (const float*)d_in[25 + i];
        va[i] = (const float*)d_in[31 + i];
    }
    const float* last_w = (const float*)d_in[37];
    const float* last_b = (const float*)d_in[38];
    const float* fc_w   = (const float*)d_in[39];
    const float* fc_b   = (const float*)d_in[40];

    char* ws = (char*)d_ws;
    size_t off = 0;
    auto carve = [&](size_t bytes) -> size_t {
        size_t cur = off;
        off += (bytes + 255) & ~(size_t)255;
        return cur;
    };
    size_t wtOff[6], biasOff[6];
    for (int i = 0; i < 6; ++i) {
        wtOff[i]   = carve((size_t)9 * COUT[i] * CPAD[i] * 2);
        biasOff[i] = carve((size_t)COUT[i] * 4);
    }
    const size_t ACT_BYTES = (size_t)BATCH * 256 * 256 * 32 * 2;
    const size_t PX_BYTES  = (size_t)BATCH * 128 * 128 * 64 * 4;
    size_t actOff[2] = {carve(ACT_BYTES), carve(ACT_BYTES)};
    size_t hOff      = carve((size_t)BATCH * 256 * 256 * 64 * 4);
    size_t pxOff[2]  = {carve(PX_BYTES), carve(PX_BYTES)};
    size_t pyOff[2]  = {carve(PX_BYTES), carve(PX_BYTES)};
    size_t tOff      = carve((size_t)BATCH * 512 * 4);

    {
        int total = BATCH * 256 * 256;
        cvt_in_kernel<<<(total + 255) / 256, 256, 0, stream>>>(x, (__bf16*)(ws + actOff[0]));
    }
    for (int i = 0; i < 6; ++i) {
        int total = 9 * COUT[i] * CPAD[i];
        prep_w_kernel<<<(total + 255) / 256, 256, 0, stream>>>(
            cw[i], cb[i], gm[i], bt[i], mu[i], va[i],
            (__bf16*)(ws + wtOff[i]), (float*)(ws + biasOff[i]),
            CIN[i], CPAD[i], COUT[i]);
    }
    for (int i = 0; i < 6; ++i) {
        const int H = HIN[i], W = HIN[i];
        const __bf16* actIn  = (const __bf16*)(ws + actOff[i & 1]);
        __bf16*       actOut = (__bf16*)(ws + actOff[(i + 1) & 1]);
        float*        hbuf   = (float*)(ws + hOff);
        const __bf16* wtp    = (const __bf16*)(ws + wtOff[i]);
        const float*  bsp    = (const float*)(ws + biasOff[i]);

        const int KC = (CPAD[i] < 256) ? CPAD[i] : 256;
        const size_t ldsBytes = (size_t)64 * (KC * 2 + 16);
        int gridM  = BATCH * H * W / 128;       // 8 waves x 16 pixels per WG
        int blocks = gridM * (COUT[i] / 64);    // exact grid, no guards

        float* pxR = (float*)(ws + pxOff[(i + 1) & 1]);
        float* pyR = (float*)(ws + pyOff[(i + 1) & 1]);
        float* pxW = (float*)(ws + pxOff[i & 1]);
        float* pyW = (float*)(ws + pyOff[i & 1]);
        long totalP = (long)BATCH * (H / 2) * (W / 2) * COUT[i];
        int pBlocks = (int)((totalP + 255) / 256);
        int Cprev = (i == 0 ? COUT[0] : COUT[i - 1]);
        int first = (i == 0) ? 1 : 0;

        switch (i) {
        case 0:
            conv_wmma_kernel<256, 256, 32, 64><<<blocks, 256, ldsBytes, stream>>>(actIn, wtp, bsp, hbuf);
            pool_ipl_kernel<256, 256, 64><<<pBlocks, 256, 0, stream>>>(hbuf, pxR, pyR, actOut, pxW, pyW, Cprev, SHIFT[i], first);
            break;
        case 1:
            conv_wmma_kernel<128, 128, 64, 128><<<blocks, 256, ldsBytes, stream>>>(actIn, wtp, bsp, hbuf);
            pool_ipl_kernel<128, 128, 128><<<pBlocks, 256, 0, stream>>>(hbuf, pxR, pyR, actOut, pxW, pyW, Cprev, SHIFT[i], first);
            break;
        case 2:
            conv_wmma_kernel<64, 64, 128, 256><<<blocks, 256, ldsBytes, stream>>>(actIn, wtp, bsp, hbuf);
            pool_ipl_kernel<64, 64, 256><<<pBlocks, 256, 0, stream>>>(hbuf, pxR, pyR, actOut, pxW, pyW, Cprev, SHIFT[i], first);
            break;
        case 3:
            conv_wmma_kernel<32, 32, 256, 512><<<blocks, 256, ldsBytes, stream>>>(actIn, wtp, bsp, hbuf);
            pool_ipl_kernel<32, 32, 512><<<pBlocks, 256, 0, stream>>>(hbuf, pxR, pyR, actOut, pxW, pyW, Cprev, SHIFT[i], first);
            break;
        case 4:
            conv_wmma_kernel<16, 16, 512, 512><<<blocks, 256, ldsBytes, stream>>>(actIn, wtp, bsp, hbuf);
            pool_ipl_kernel<16, 16, 512><<<pBlocks, 256, 0, stream>>>(hbuf, pxR, pyR, actOut, pxW, pyW, Cprev, SHIFT[i], first);
            break;
        default:
            conv_wmma_kernel<8, 8, 512, 512><<<blocks, 256, ldsBytes, stream>>>(actIn, wtp, bsp, hbuf);
            pool_ipl_kernel<8, 8, 512><<<pBlocks, 256, 0, stream>>>(hbuf, pxR, pyR, actOut, pxW, pyW, Cprev, SHIFT[i], first);
            break;
        }
    }
    float* pxF = (float*)(ws + pxOff[1]);
    float* pyF = (float*)(ws + pyOff[1]);
    head1_kernel<<<(BATCH * 512 + 255) / 256, 256, 0, stream>>>(
        pxF, pyF, last_w, last_b, (float*)(ws + tOff));
    head2_kernel<<<BATCH, 256, 0, stream>>>(
        (float*)(ws + tOff), fc_w, fc_b, (float*)d_out);
}
